// MultiHeadAttention_2714419331128
// MI455X (gfx1250) — matrix operand order repaired and checked by reading, not compiled, not run on hardware
//
#include <hip/hip_runtime.h>
#include <hip/hip_bf16.h>
#include <math.h>

// Problem constants (B=2, S=2048, D=1024, H=16, HD=64)
#define BATCH  2
#define SEQ    2048
#define DMODEL 1024
#define NHEAD  16
#define HEADD  64
#define ROWS   (BATCH * SEQ)   // 4096

typedef __attribute__((ext_vector_type(16))) __bf16 v16bf;
typedef __attribute__((ext_vector_type(8)))  float  v8f;

union BfU { unsigned short u; __bf16 b; };

__device__ __forceinline__ unsigned short f2bf(float f) {
    union { float f; unsigned u; } v; v.f = f;
    unsigned r = (v.u + 0x7FFFu + ((v.u >> 16) & 1u)) >> 16;   // RNE
    return (unsigned short)r;
}
__device__ __forceinline__ __bf16 us2bf(unsigned short s) { BfU t; t.u = s; return t.b; }

__device__ __forceinline__ void wave_lds_fence() {
#if __has_builtin(__builtin_amdgcn_s_wait_dscnt)
    __builtin_amdgcn_s_wait_dscnt(0);
#else
    asm volatile("s_wait_dscnt 0x0" ::: "memory");
#endif
}

// Fragment swizzle maps (CDNA5 ISA §7.12.2):
//  A-frag (16x32):  K = h + (h>=8)*8 + (lane>=16)*8,  M = lane%16
//    inverse: lanegrp = (K>>3)&1, h = (K&7) + ((K&16)?8:0)
//  B-frag (32x16):  K = h + (h>=8)*8 + (lane>=16)*8,            N = lane%16
//    inverse: lanegrp = (K>>3)&1, h = (K&7) + ((K&16)?8:0)
// Swizzled fragment storage: frag*512 + lane*16 + h  (halves) -> lane reads
// 32 contiguous bytes (2x ds_load_b128 / global_load_b128).

// ---------------------------------------------------------------------------
// Kernel 1a: fp32 -> bf16, 8 elements/thread, b128 in / b128 out
// ---------------------------------------------------------------------------
__global__ __launch_bounds__(256)
void cvt_f32_to_bf16_x8(const float* __restrict__ src,
                        unsigned short* __restrict__ dst, int n8) {
    int i = blockIdx.x * blockDim.x + threadIdx.x;
    if (i >= n8) return;
    const float4* s = reinterpret_cast<const float4*>(src) + (size_t)i * 2;
    float4 a = s[0], b = s[1];
    unsigned short o[8] __attribute__((aligned(16)));
    o[0] = f2bf(a.x); o[1] = f2bf(a.y); o[2] = f2bf(a.z); o[3] = f2bf(a.w);
    o[4] = f2bf(b.x); o[5] = f2bf(b.y); o[6] = f2bf(b.z); o[7] = f2bf(b.w);
    reinterpret_cast<uint4*>(dst)[i] = *reinterpret_cast<uint4*>(o);
}

// ---------------------------------------------------------------------------
// Kernel 1b: fp32 [k][n] -> bf16 transposed [n][k], LDS-tiled 64x64
// ---------------------------------------------------------------------------
__global__ __launch_bounds__(256)
void cvt_transpose_bf16(const float* __restrict__ src,
                        unsigned short* __restrict__ dst) {
    __shared__ unsigned short tile[64][72];   // pad to dodge bank conflicts
    const int k0 = blockIdx.x * 64;
    const int n0 = blockIdx.y * 64;
    const int t  = threadIdx.x;
    const int rr = t >> 2;             // 0..63
    const int cb = (t & 3) * 16;       // 0/16/32/48
    for (int j = 0; j < 16; ++j)
        tile[rr][cb + j] = f2bf(src[(size_t)(k0 + rr) * DMODEL + n0 + cb + j]);
    __syncthreads();
    unsigned short ob[16] __attribute__((aligned(16)));
    for (int j = 0; j < 16; ++j) ob[j] = tile[cb + j][rr];
    uint4* gd = reinterpret_cast<uint4*>(&dst[(size_t)(n0 + rr) * DMODEL + k0 + cb]);
    gd[0] = *reinterpret_cast<uint4*>(&ob[0]);
    gd[1] = *reinterpret_cast<uint4*>(&ob[8]);
}

// ---------------------------------------------------------------------------
// Kernel 2: bf16 WMMA GEMM, C = A[Mx1024] * W + bias, W given transposed
//   [n][k]. WG tile 128x128, 8 waves (4 M-slabs x 2 N-slabs), wave 32x64.
//   LDS tiles stored fragment-swizzled: all frag loads are ds_load_b128.
//   MODE is a template parameter -> one specialized epilogue per
//   instantiation (no live dead-branch address math, no acc spills).
//   MODE 0: Q -> global fragment-swizzled bf16   [bh][s/16][kc][lane][h]
//   MODE 1: K -> bf16 [B,H,S,HD]
//   MODE 2: V -> bf16 [B,H,HD,S]   (transposed for b128-clean attn staging)
//   MODE 3: f32 [M x 1024] (output projection)
// ---------------------------------------------------------------------------
template <int MODE>
__global__ __launch_bounds__(256)
void gemm_bf16_wmma(const unsigned short* __restrict__ A,
                    const unsigned short* __restrict__ Wt,   // [n][k] bf16
                    const float* __restrict__ bias,
                    unsigned short* __restrict__ out_bf16,
                    float* __restrict__ out_f32) {
    __shared__ unsigned short As[8 * 512];   // 8 M-subtile fragments
    __shared__ unsigned short Bs[8 * 512];   // 8 N-subtile fragments

    const int tid  = threadIdx.x;
    const int lane = tid & 31;
    const int wave = tid >> 5;
    const int wm   = wave & 3;
    const int wn   = wave >> 2;
    const int m0   = blockIdx.y * 128;
    const int n0   = blockIdx.x * 128;
    const int lrow = lane & 15;

    v8f acc[2][4];
    for (int mi = 0; mi < 2; ++mi)
        for (int ni = 0; ni < 4; ++ni)
            for (int r = 0; r < 8; ++r) acc[mi][ni][r] = 0.f;

    for (int kt = 0; kt < DMODEL; kt += 32) {
        // ---- stage A (128x32) swizzled: 512 uint4, 2 per thread ----
        for (int rep = 0; rep < 2; ++rep) {
            int idx = tid + rep * 256;
            int row = idx >> 2, c = idx & 3;            // k-chunk of 8
            const uint4 ga = *reinterpret_cast<const uint4*>(
                &A[(size_t)(m0 + row) * DMODEL + kt + c * 8]);
            int lg = c & 1, h0 = (c >> 1) * 8;          // A-frag inverse map
            int l  = (row & 15) + lg * 16;
            *reinterpret_cast<uint4*>(&As[(size_t)(row >> 4) * 512 + l * 16 + h0]) = ga;
        }
        // ---- stage B (32x128) swizzled from Wt rows: 512 uint4 ----
        for (int rep = 0; rep < 2; ++rep) {
            int idx = tid + rep * 256;
            int n = idx >> 2, c = idx & 3;              // k-chunk of 8
            const uint4 gb = *reinterpret_cast<const uint4*>(
                &Wt[(size_t)(n0 + n) * DMODEL + kt + c * 8]);
            int lg = c & 1, h0 = (c >> 1) * 8;          // B-frag inverse map
            int l  = (n & 15) + lg * 16;
            *reinterpret_cast<uint4*>(&Bs[(size_t)(n >> 4) * 512 + l * 16 + h0]) = gb;
        }
        if (kt + 32 < DMODEL) {   // near (WGP-scope) prefetch of next K tile
            __builtin_prefetch(&A[(size_t)(m0 + (tid >> 2)) * DMODEL + kt + 32], 0, 3);
            __builtin_prefetch(&Wt[(size_t)(n0 + (tid >> 2)) * DMODEL + kt + 32], 0, 3);
        }
        __syncthreads();

        v16bf af[2], bfr[4];
        for (int mi = 0; mi < 2; ++mi)
            af[mi] = *reinterpret_cast<const v16bf*>(&As[(size_t)(wm * 2 + mi) * 512 + lane * 16]);
        for (int ni = 0; ni < 4; ++ni)
            bfr[ni] = *reinterpret_cast<const v16bf*>(&Bs[(size_t)(wn * 4 + ni) * 512 + lane * 16]);

        for (int mi = 0; mi < 2; ++mi)
            for (int ni = 0; ni < 4; ++ni)
                acc[mi][ni] = __builtin_amdgcn_wmma_f32_16x16x32_bf16(
                    false, af[mi], false, bfr[ni], (short)0, acc[mi][ni], false, false);
        __syncthreads();
    }

    // Epilogue. C/D layout: VGPR r -> M = r + (lane>=16)*8, N = lane%16
    for (int mi = 0; mi < 2; ++mi) {
        for (int ni = 0; ni < 4; ++ni) {
            const int col = n0 + wn * 64 + ni * 16 + lrow;
            const float bv = bias[col];
            for (int r = 0; r < 8; ++r) {
                const int row = m0 + wm * 32 + mi * 16 + r + ((lane & 16) ? 8 : 0);
                const float vout = acc[mi][ni][r] + bv;
                if (MODE == 3) {
                    out_f32[(size_t)row * DMODEL + col] = vout;
                } else {
                    const int b = row >> 11, s = row & (SEQ - 1);
                    const int h = col >> 6,  d = col & (HEADD - 1);
                    const int bh = b * NHEAD + h;
                    if (MODE == 0) {           // Q: global fragment-swizzled
                        const int kc = d >> 5, k32 = d & 31;
                        const int lg = (k32 >> 3) & 1;
                        const int hh = (k32 & 7) + ((k32 & 16) ? 8 : 0);
                        const int l  = (s & 15) + lg * 16;
                        const size_t qidx =
                            (((size_t)bh * 128 + (s >> 4)) * 2 + kc) * 512 + l * 16 + hh;
                        out_bf16[qidx] = f2bf(vout);
                    } else if (MODE == 1) {    // K: [B,H,S,HD]
                        out_bf16[((size_t)bh * SEQ + s) * HEADD + d] = f2bf(vout);
                    } else {                   // V: [B,H,HD,S]
                        out_bf16[((size_t)bh * HEADD + d) * SEQ + s] = f2bf(vout);
                    }
                }
            }
        }
    }
}

// ---------------------------------------------------------------------------
// Kernel 3: flash attention. 4 waves x 16 query rows per WG, 64-key blocks.
//   K/V staged fragment-swizzled (b128 stores); all frag loads ds_load_b128.
// ---------------------------------------------------------------------------
__global__ __launch_bounds__(128)
void attn_flash_wmma(const unsigned short* __restrict__ Qsw,  // frag-swizzled
                     const unsigned short* __restrict__ Kb,   // [B,H,S,HD]
                     const unsigned short* __restrict__ Vt,   // [B,H,HD,S]
                     const float* __restrict__ amask,         // [B, S]
                     unsigned short* __restrict__ ctx) {      // [B,S,D] bf16
    __shared__ unsigned short Ksw[8 * 512];   // frag (ksub*2+ks)
    __shared__ unsigned short Vsw[8 * 512];   // frag (ks*4+nsub)
    __shared__ unsigned short Psw[4 * 1024];  // per-wave, frag kc

    const int tid  = threadIdx.x;
    const int lane = tid & 31;
    const int wave = tid >> 5;
    const int q0   = blockIdx.x * 64;
    const int bh   = blockIdx.y;
    const int b    = bh >> 4;
    const int h    = bh & 15;
    const size_t headbase = (size_t)bh * SEQ * HEADD;

    const int lrow = lane & 15;
    const int hi8  = (lane & 16) ? 8 : 0;

    // Q fragments: contiguous global_load_b128 pairs (pre-swizzled by GEMM)
    v16bf qf[2];
    {
        const size_t sub = (size_t)bh * 128 + (q0 >> 4) + wave;
        qf[0] = *reinterpret_cast<const v16bf*>(&Qsw[(sub * 2 + 0) * 512 + lane * 16]);
        qf[1] = *reinterpret_cast<const v16bf*>(&Qsw[(sub * 2 + 1) * 512 + lane * 16]);
    }

    float mrun[8], lrun[8];
    v8f o[4];
    for (int r = 0; r < 8; ++r) { mrun[r] = -1e30f; lrun[r] = 0.f; }
    for (int ni = 0; ni < 4; ++ni)
        for (int r = 0; r < 8; ++r) o[ni][r] = 0.f;

    for (int kb = 0; kb < SEQ; kb += 64) {
        // ---- stage K (64 keys x 64 hd) swizzled: 512 uint4, 4/thread ----
        for (int rep = 0; rep < 4; ++rep) {
            int idx = tid + rep * 128;
            int key = idx >> 3, c = idx & 7;            // hd-chunk of 8
            const uint4 gk = *reinterpret_cast<const uint4*>(
                &Kb[headbase + (size_t)(kb + key) * HEADD + c * 8]);
            int ks = c >> 2, c2 = c & 3;
            int lg = c2 & 1, h0 = (c2 >> 1) * 8;        // B-frag inverse map
            int l  = (key & 15) + lg * 16;
            int frag = (key >> 4) * 2 + ks;
            *reinterpret_cast<uint4*>(&Ksw[(size_t)frag * 512 + l * 16 + h0]) = gk;
        }
        // ---- stage V (64 hd x 64 keys, from transposed Vt) swizzled ----
        for (int rep = 0; rep < 4; ++rep) {
            int idx = tid + rep * 128;
            int hd = idx >> 3, c = idx & 7;             // key-chunk of 8
            const uint4 gv = *reinterpret_cast<const uint4*>(
                &Vt[headbase + (size_t)hd * SEQ + kb + c * 8]);
            int ks = c >> 2, c2 = c & 3;
            int lg = c2 & 1, h0 = (c2 >> 1) * 8;
            int l  = (hd & 15) + lg * 16;
            int frag = ks * 4 + (hd >> 4);
            *reinterpret_cast<uint4*>(&Vsw[(size_t)frag * 512 + l * 16 + h0]) = gv;
        }
        __syncthreads();

        // ---- S = Q K^T (4 tiles of 16 keys, hd contraction in 2 steps) ----
        v8f sc[4];
        for (int ni = 0; ni < 4; ++ni) {
            v8f c; for (int r = 0; r < 8; ++r) c[r] = 0.f;
            for (int ks = 0; ks < 2; ++ks) {
                v16bf kf = *reinterpret_cast<const v16bf*>(
                    &Ksw[(size_t)(ni * 2 + ks) * 512 + lane * 16]);
                c = __builtin_amdgcn_wmma_f32_16x16x32_bf16(
                        false, qf[ks], false, kf, (short)0, c, false, false);
            }
            const float msk = amask[b * SEQ + kb + ni * 16 + lrow];
            for (int r = 0; r < 8; ++r) sc[ni][r] = c[r] * 0.125f + msk;
        }

        // ---- online softmax (rows on 16-lane groups, xor-reduce) ----
        float alpha[8];
        for (int r = 0; r < 8; ++r) {
            float mx = fmaxf(fmaxf(sc[0][r], sc[1][r]), fmaxf(sc[2][r], sc[3][r]));
            mx = fmaxf(mx, __shfl_xor(mx, 1));
            mx = fmaxf(mx, __shfl_xor(mx, 2));
            mx = fmaxf(mx, __shfl_xor(mx, 4));
            mx = fmaxf(mx, __shfl_xor(mx, 8));
            float mn = fmaxf(mrun[r], mx);
            alpha[r] = __expf(mrun[r] - mn);
            mrun[r]  = mn;
            float rs = 0.f;
            for (int ni = 0; ni < 4; ++ni) {
                float p = __expf(sc[ni][r] - mn);
                sc[ni][r] = p;
                rs += p;
            }
            rs += __shfl_xor(rs, 1);
            rs += __shfl_xor(rs, 2);
            rs += __shfl_xor(rs, 4);
            rs += __shfl_xor(rs, 8);
            lrun[r] = lrun[r] * alpha[r] + rs;
        }
        for (int ni = 0; ni < 4; ++ni)
            for (int r = 0; r < 8; ++r) o[ni][r] *= alpha[r];

        // ---- P: D-layout -> A-layout via per-wave swizzled LDS scratch ----
        {
            const int lg = (lrow >> 3) & 1;             // from key low bits
            for (int ni = 0; ni < 4; ++ni) {
                const int kc = ni >> 1;
                const int hh = (lrow & 7) + ((ni & 1) ? 8 : 0);
                for (int r = 0; r < 8; ++r) {
                    int l = (r + hi8) + lg * 16;
                    Psw[(size_t)wave * 1024 + kc * 512 + l * 16 + hh] = f2bf(sc[ni][r]);
                }
            }
        }
        wave_lds_fence();   // s_wait_dscnt 0 — wave-local LDS RAW guard

        v16bf pf[2];
        pf[0] = *reinterpret_cast<const v16bf*>(&Psw[(size_t)wave * 1024 + lane * 16]);
        pf[1] = *reinterpret_cast<const v16bf*>(&Psw[(size_t)wave * 1024 + 512 + lane * 16]);

        // ---- O += P V (key contraction in 2 steps, 4 hd tiles) ----
        for (int ni = 0; ni < 4; ++ni)
            for (int ks = 0; ks < 2; ++ks) {
                v16bf vf = *reinterpret_cast<const v16bf*>(
                    &Vsw[(size_t)(ks * 4 + ni) * 512 + lane * 16]);
                o[ni] = __builtin_amdgcn_wmma_f32_16x16x32_bf16(
                            false, pf[ks], false, vf, (short)0, o[ni], false, false);
            }
        __syncthreads();
    }

    // normalize, merge heads to [B,S,D] bf16
    for (int ni = 0; ni < 4; ++ni)
        for (int r = 0; r < 8; ++r) {
            int s = q0 + wave * 16 + r + hi8;
            float vv = o[ni][r] / lrun[r];
            ctx[((size_t)(b * SEQ + s)) * DMODEL + h * HEADD + ni * 16 + lrow] = f2bf(vv);
        }
}

// ---------------------------------------------------------------------------
// Kernel 4: residual add + LayerNorm (one row per block)
// ---------------------------------------------------------------------------
__global__ __launch_bounds__(256)
void resid_layernorm(const float* __restrict__ hidden,
                     const float* __restrict__ inp,
                     const float* __restrict__ gamma,
                     const float* __restrict__ beta,
                     float* __restrict__ out) {
    __shared__ float s_sum[8], s_sq[8];
    const int row  = blockIdx.x;
    const int tid  = threadIdx.x;
    const int lane = tid & 31, wave = tid >> 5;
    float x[4];
    float sum = 0.f, sq = 0.f;
    for (int k = 0; k < 4; ++k) {
        int j = tid + k * 256;
        x[k] = hidden[(size_t)row * DMODEL + j] + inp[(size_t)row * DMODEL + j];
        sum += x[k]; sq += x[k] * x[k];
    }
    for (int m = 1; m < 32; m <<= 1) { sum += __shfl_xor(sum, m); sq += __shfl_xor(sq, m); }
    if (lane == 0) { s_sum[wave] = sum; s_sq[wave] = sq; }
    __syncthreads();
    sum = 0.f; sq = 0.f;
    for (int w = 0; w < 8; ++w) { sum += s_sum[w]; sq += s_sq[w]; }
    float mu   = sum * (1.f / DMODEL);
    float var  = sq * (1.f / DMODEL) - mu * mu;
    float rstd = rsqrtf(var + 1e-12f);
    for (int k = 0; k < 4; ++k) {
        int j = tid + k * 256;
        out[(size_t)row * DMODEL + j] = (x[k] - mu) * rstd * gamma[j] + beta[j];
    }
}

// ---------------------------------------------------------------------------
// Orchestration
// ---------------------------------------------------------------------------
extern "C" void kernel_launch(void* const* d_in, const int* in_sizes, int n_in,
                              void* d_out, int out_size, void* d_ws, size_t ws_size,
                              hipStream_t stream) {
    (void)in_sizes; (void)n_in; (void)out_size; (void)ws_size;
    const float* x     = (const float*)d_in[0];
    const float* am    = (const float*)d_in[1];
    const float* Wq    = (const float*)d_in[2];
    const float* bq    = (const float*)d_in[3];
    const float* Wk    = (const float*)d_in[4];
    const float* bk    = (const float*)d_in[5];
    const float* Wv    = (const float*)d_in[6];
    const float* bv    = (const float*)d_in[7];
    const float* Wo    = (const float*)d_in[8];
    const float* bo    = (const float*)d_in[9];
    const float* gamma = (const float*)d_in[10];
    const float* beta  = (const float*)d_in[11];
    float* out = (float*)d_out;

    const size_t nX = (size_t)ROWS * DMODEL;     // 4096*1024
    const size_t nW = (size_t)DMODEL * DMODEL;   // 1024*1024

    char* ws = (char*)d_ws;
    unsigned short* Xb   = (unsigned short*)ws;  ws += nX * 2;
    unsigned short* Wqt  = (unsigned short*)ws;  ws += nW * 2;
    unsigned short* Wkt  = (unsigned short*)ws;  ws += nW * 2;
    unsigned short* Wvt  = (unsigned short*)ws;  ws += nW * 2;
    unsigned short* Wot  = (unsigned short*)ws;  ws += nW * 2;
    unsigned short* Qsw  = (unsigned short*)ws;  ws += nX * 2;
    unsigned short* Kbf  = (unsigned short*)ws;  ws += nX * 2;
    unsigned short* Vtb  = (unsigned short*)ws;  ws += nX * 2;
    unsigned short* Ctxb = (unsigned short*)ws;  ws += nX * 2;
    float*          Hid  = (float*)ws;           ws += nX * 4;

    // 1) fp32 -> bf16 (activations straight; weights transposed to [n][k])
    cvt_f32_to_bf16_x8<<<(unsigned)(nX / (256 * 8)), 256, 0, stream>>>(x, Xb, (int)(nX / 8));
    dim3 tgrid(DMODEL / 64, DMODEL / 64);
    cvt_transpose_bf16<<<tgrid, 256, 0, stream>>>(Wq, Wqt);
    cvt_transpose_bf16<<<tgrid, 256, 0, stream>>>(Wk, Wkt);
    cvt_transpose_bf16<<<tgrid, 256, 0, stream>>>(Wv, Wvt);
    cvt_transpose_bf16<<<tgrid, 256, 0, stream>>>(Wo, Wot);

    // 2) QKV projections (WMMA); Q frag-swizzled, K row-major, V transposed
    dim3 ggrid(DMODEL / 128, ROWS / 128);   // (8, 32)
    gemm_bf16_wmma<0><<<ggrid, 256, 0, stream>>>(Xb, Wqt, bq, Qsw, nullptr);
    gemm_bf16_wmma<1><<<ggrid, 256, 0, stream>>>(Xb, Wkt, bk, Kbf, nullptr);
    gemm_bf16_wmma<2><<<ggrid, 256, 0, stream>>>(Xb, Wvt, bv, Vtb, nullptr);

    // 3) flash attention (WMMA QK^T + PV), merges heads to [B,S,D] bf16
    attn_flash_wmma<<<dim3(SEQ / 64, BATCH * NHEAD), 128, 0, stream>>>(Qsw, Kbf, Vtb, am, Ctxb);

    // 4) output projection (WMMA) -> f32 hidden
    gemm_bf16_wmma<3><<<ggrid, 256, 0, stream>>>(Ctxb, Wot, bo, nullptr, Hid);

    // 5) residual + LayerNorm -> d_out
    resid_layernorm<<<ROWS, 256, 0, stream>>>(Hid, x, gamma, beta, out);
}
